// SelfAttention_58909771432626
// MI455X (gfx1250) — matrix operand order repaired and checked by reading, not compiled, not run on hardware
//
#include <hip/hip_runtime.h>

// ---------------------------------------------------------------------------
// Self-attention forward for MI455X (gfx1250, wave32, WMMA).
// All GEMMs run on v_wmma_f32_16x16x32_bf16 with fp32 accumulation.
// Attention K/V tiles staged with GLOBAL_LOAD_ASYNC_TO_LDS (ASYNCcnt).
// ---------------------------------------------------------------------------

typedef __bf16 bf16;
typedef __attribute__((ext_vector_type(8)))  __bf16 bf16x8;
typedef __attribute__((ext_vector_type(16))) __bf16 bf16x16;
typedef __attribute__((ext_vector_type(8)))  float  f32x8;
typedef int v4i __attribute__((vector_size(16)));  // int4, matches builtin sig

#define HID   1024
#define SEQ   2048
#define BATCH 4
#define NHEAD 16
#define HDIM  64

static __device__ __forceinline__ bf16x16 bfcomb(bf16x8 lo, bf16x8 hi) {
  return __builtin_shufflevector(lo, hi, 0,1,2,3,4,5,6,7,8,9,10,11,12,13,14,15);
}

static __device__ __forceinline__ f32x8 wmma_bf16(bf16x16 a, bf16x16 b, f32x8 c) {
  // (neg_a, A, neg_b, B, c_mod, C, reuse_a, reuse_b)
  return __builtin_amdgcn_wmma_f32_16x16x32_bf16(false, a, false, b, (short)0, c,
                                                 false, false);
}

// ---- async global -> LDS copy of 16 bytes per lane (ASYNCcnt tracked) -----
#if __has_builtin(__builtin_amdgcn_global_load_async_to_lds_b128)
static __device__ __forceinline__ void async_cp16(const bf16* src, bf16* dst_lds) {
  // Builtin signature: (v4i addrspace(1)* src, v4i addrspace(3)* dst,
  //                     imm offset, imm cpol)
  __builtin_amdgcn_global_load_async_to_lds_b128(
      (__attribute__((address_space(1))) v4i*)(v4i*)(void*)const_cast<bf16*>(src),
      (__attribute__((address_space(3))) v4i*)(v4i*)(void*)dst_lds,
      0, 0);
}
#else
static __device__ __forceinline__ void async_cp16(const bf16* src, bf16* dst_lds) {
  // VDST VGPR holds the per-lane LDS byte offset; VADDR holds the 64-bit
  // global address (GV mode).
  asm volatile("global_load_async_to_lds_b128 %0, %1, off"
               :: "v"((unsigned)(size_t)dst_lds),
                  "v"((unsigned long long)(size_t)src)
               : "memory");
}
#endif

static __device__ __forceinline__ void wait_async0() {
#if __has_builtin(__builtin_amdgcn_s_wait_asynccnt)
  __builtin_amdgcn_s_wait_asynccnt(0);
#else
  asm volatile("s_wait_asynccnt 0x0" ::: "memory");
#endif
}

// ---------------------------------------------------------------------------
// Generic GEMM:  C[M,N] = A[M,K] * W[K,N] + bias[N]
//   A: fp32 or bf16 (template); W,bias: fp32.
//   OUT_MODE: 0 = fp32 [M,N]; 1 = bf16 [M,N];
//             2 = bf16 transposed per head: [B, H, D, S] (for the V path).
// Block tile 128x128, K-tile 32, 256 threads = 8 waves.
// Wave (wm in 0..3, wn in 0..1) computes a 32x64 sub-tile = 2x4 WMMA tiles.
// A staged row-major bf16 [m][k] (stride 40); W staged transposed [n][k].
// ---------------------------------------------------------------------------
template <bool A_BF16, int OUT_MODE>
__global__ __launch_bounds__(256)
void gemm_bias_kernel(const void* __restrict__ Av, const float* __restrict__ W,
                      const float* __restrict__ bias, void* __restrict__ Cv,
                      int M, int N, int K) {
  constexpr int ASTR = 40;  // 80 B row stride (16B multiple, bank-phase offset)
  constexpr int WSTR = 40;
  __shared__ bf16 lds_a[128 * ASTR];
  __shared__ bf16 lds_w[128 * WSTR];

  const int tid  = threadIdx.x;
  const int lane = tid & 31;
  const int wave = tid >> 5;
  const int l16  = lane & 15;
  const int half = lane >> 4;
  const int wm   = wave & 3;   // 4 m-tiles of 32 rows
  const int wn   = wave >> 2;  // 2 n-tiles of 64 cols
  const int m0   = blockIdx.x * 128;
  const int n0   = blockIdx.y * 128;

  const f32x8 fz = {0.f,0.f,0.f,0.f,0.f,0.f,0.f,0.f};
  f32x8 acc[2][4];
#pragma unroll
  for (int mi = 0; mi < 2; ++mi)
#pragma unroll
    for (int ni = 0; ni < 4; ++ni) acc[mi][ni] = fz;

  const int tm  = tid >> 2;         // 0..63 : A row pair owner
  const int tk0 = (tid & 3) * 8;    // 0,8,16,24
  const int tn  = tid & 63;         // 0..63 : W column pair owner
  const int tkw = (tid >> 6) * 8;   // 0,8,16,24

  for (int kt = 0; kt < K; kt += 32) {
    // ---- stage A tile (128x32 -> bf16 LDS) -------------------------------
#pragma unroll
    for (int r = 0; r < 2; ++r) {
      const int m = tm + r * 64;
      bf16x8 v;
      if (A_BF16) {
        v = *(const bf16x8*)((const bf16*)Av + (size_t)(m0 + m) * K + kt + tk0);
      } else {
        const float* ap = (const float*)Av + (size_t)(m0 + m) * K + kt + tk0;
        if (kt + 32 < K) __builtin_prefetch(ap + 32, 0, 0);  // global_prefetch_b8
#pragma unroll
        for (int i = 0; i < 8; ++i) v[i] = (bf16)ap[i];
      }
      *(bf16x8*)&lds_a[m * ASTR + tk0] = v;
    }
    // ---- stage W tile transposed (lds_w[n][k]), vectorized stores --------
    {
      bf16x8 w0, w1;
#pragma unroll
      for (int i = 0; i < 8; ++i) {
        const int kk = kt + tkw + i;
        w0[i] = (bf16)W[(size_t)kk * N + n0 + tn];
        w1[i] = (bf16)W[(size_t)kk * N + n0 + tn + 64];
      }
      *(bf16x8*)&lds_w[tn * WSTR + tkw]        = w0;
      *(bf16x8*)&lds_w[(tn + 64) * WSTR + tkw] = w1;
    }
    __syncthreads();

    // ---- fragments + 8 WMMAs ---------------------------------------------
    bf16x16 af[2];
#pragma unroll
    for (int mi = 0; mi < 2; ++mi) {
      // A 16x32 bf16 layout: lane row = l16; k = 8*half..+7 and 16+8*half..+7
      const bf16* p = &lds_a[(wm * 32 + mi * 16 + l16) * ASTR + 8 * half];
      af[mi] = bfcomb(*(const bf16x8*)p, *(const bf16x8*)(p + 16));
    }
#pragma unroll
    for (int ni = 0; ni < 4; ++ni) {
      // B 32x16 bf16 layout: lane col = l16; k = 8*half..+7 and 16+8*half..+7
      const bf16* p = &lds_w[(wn * 64 + ni * 16 + l16) * WSTR + 8 * half];
      const bf16x16 bv = bfcomb(*(const bf16x8*)p, *(const bf16x8*)(p + 16));
#pragma unroll
      for (int mi = 0; mi < 2; ++mi)
        acc[mi][ni] = wmma_bf16(af[mi], bv, acc[mi][ni]);
    }
    __syncthreads();
  }

  // ---- epilogue: bias + store (C layout: row = j + 8*half, col = l16) ----
#pragma unroll
  for (int ni = 0; ni < 4; ++ni) {
    const int n = n0 + wn * 64 + ni * 16 + l16;
    const float bn = bias[n];
#pragma unroll
    for (int mi = 0; mi < 2; ++mi) {
#pragma unroll
      for (int j = 0; j < 8; ++j) {
        const int m = m0 + wm * 32 + mi * 16 + j + 8 * half;
        const float v = acc[mi][ni][j] + bn;
        if (OUT_MODE == 0) {
          ((float*)Cv)[(size_t)m * N + n] = v;
        } else if (OUT_MODE == 1) {
          ((bf16*)Cv)[(size_t)m * N + n] = (bf16)v;
        } else {
          // [B, H, D, S]: key-contiguous rows for the attention V operand
          const int bb = m >> 11;        // m / SEQ
          const int ss = m & (SEQ - 1);  // m % SEQ
          const int hh = n >> 6;         // n / HDIM
          const int dd = n & (HDIM - 1); // n % HDIM
          ((bf16*)Cv)[((size_t)(bb * NHEAD + hh) * HDIM + dd) * SEQ + ss] =
              (bf16)v;
        }
      }
    }
  }
}

// ---------------------------------------------------------------------------
// Flash attention: block = (b, h, 128 query rows); 8 waves x 16 rows each.
// Q,Ctx: bf16 [B, S, HID] (head h in columns h*64..h*64+63); K: same layout;
// V: bf16 [B, H, D, S] (pre-transposed by its projection GEMM).
// K/V tiles are staged into LDS with async global->LDS b128 copies.
// Online softmax runs on the WMMA C-layout: a row lives at a fixed VGPR
// index (j + 8*half) broadcast across 16 lanes, so row max/sum are 16-lane
// __shfl_xor reductions and rescale is a per-element multiply.
// ---------------------------------------------------------------------------
__global__ __launch_bounds__(256)
void attn_kernel(const bf16* __restrict__ Q, const bf16* __restrict__ Kw,
                 const bf16* __restrict__ Vt, bf16* __restrict__ Ctx) {
  constexpr int KSTR = 72;  // 144 B rows, 16B multiple
  __shared__ bf16 lds_k[64 * KSTR];        // [key][d]
  __shared__ bf16 lds_v[64 * KSTR];        // [d][key]
  __shared__ bf16 lds_p[8 * 16 * KSTR];    // per-wave P relayout buffer

  const int tid  = threadIdx.x;
  const int lane = tid & 31;
  const int wave = tid >> 5;
  const int l16  = lane & 15;
  const int half = lane >> 4;

  const int b  = blockIdx.z;
  const int h  = blockIdx.y;
  const int q0 = blockIdx.x * 128 + wave * 16;
  const size_t bh  = (size_t)b * SEQ * HID + (size_t)h * HDIM;
  const size_t vhb = (size_t)(b * NHEAD + h) * HDIM * SEQ;

  // ---- Q fragments for this wave's 16 rows (two K-chunks of 32) ----------
  bf16x16 qf[2];
  {
    const bf16* qp = Q + bh + (size_t)(q0 + l16) * HID;
#pragma unroll
    for (int c = 0; c < 2; ++c) {
      const int d0 = c * 32 + 8 * half;
      qf[c] = bfcomb(*(const bf16x8*)(qp + d0), *(const bf16x8*)(qp + d0 + 16));
    }
  }

  const f32x8 fz = {0.f,0.f,0.f,0.f,0.f,0.f,0.f,0.f};
  f32x8 oacc[4];
#pragma unroll
  for (int dt = 0; dt < 4; ++dt) oacc[dt] = fz;
  float mrow[8], lrow[8];
#pragma unroll
  for (int j = 0; j < 8; ++j) { mrow[j] = -3.0e38f; lrow[j] = 0.f; }

  bf16* pbuf = lds_p + wave * 16 * KSTR;
  const int srow = tid >> 2;          // staging row: key (for K) / d (for V)
  const int sc0  = (tid & 3) * 16;    // staging col: d (for K) / key (for V)

  for (int k0 = 0; k0 < SEQ; k0 += 64) {
    __syncthreads();  // previous tile fully consumed before overwrite
    // ---- async-stage K [key][d] and V [d][key] ---------------------------
    {
      const bf16* kp = Kw + bh + (size_t)(k0 + srow) * HID + sc0;
      async_cp16(kp,     &lds_k[srow * KSTR + sc0]);
      async_cp16(kp + 8, &lds_k[srow * KSTR + sc0 + 8]);
      const bf16* vp = Vt + vhb + (size_t)srow * SEQ + k0 + sc0;
      async_cp16(vp,     &lds_v[srow * KSTR + sc0]);
      async_cp16(vp + 8, &lds_v[srow * KSTR + sc0 + 8]);
    }
    wait_async0();
    __syncthreads();

    // ---- S = Q * K^T  (16 q-rows x 64 keys) ------------------------------
    f32x8 sacc[4];
#pragma unroll
    for (int kt = 0; kt < 4; ++kt) sacc[kt] = fz;
#pragma unroll
    for (int kt = 0; kt < 4; ++kt) {
#pragma unroll
      for (int c = 0; c < 2; ++c) {
        const bf16* p = &lds_k[(kt * 16 + l16) * KSTR + c * 32 + 8 * half];
        const bf16x16 bv = bfcomb(*(const bf16x8*)p, *(const bf16x8*)(p + 16));
        sacc[kt] = wmma_bf16(qf[c], bv, sacc[kt]);
      }
    }

    // ---- online softmax (scale 1/sqrt(64) = 0.125) -----------------------
#pragma unroll
    for (int kt = 0; kt < 4; ++kt)
#pragma unroll
      for (int j = 0; j < 8; ++j) sacc[kt][j] *= 0.125f;

    float mt[8];
#pragma unroll
    for (int j = 0; j < 8; ++j) {
      float v = sacc[0][j];
#pragma unroll
      for (int kt = 1; kt < 4; ++kt) v = fmaxf(v, sacc[kt][j]);
#pragma unroll
      for (int mask = 1; mask <= 8; mask <<= 1)
        v = fmaxf(v, __shfl_xor(v, mask, 32));
      mt[j] = v;
    }
    float al[8], rs[8];
#pragma unroll
    for (int j = 0; j < 8; ++j) {
      const float mn = fmaxf(mrow[j], mt[j]);
      al[j] = __expf(mrow[j] - mn);
      mrow[j] = mn;
      rs[j] = 0.f;
    }
#pragma unroll
    for (int kt = 0; kt < 4; ++kt) {
#pragma unroll
      for (int j = 0; j < 8; ++j) {
        const float p = __expf(sacc[kt][j] - mrow[j]);
        rs[j] += p;
        pbuf[(j + 8 * half) * KSTR + kt * 16 + l16] = (bf16)p;
      }
    }
#pragma unroll
    for (int j = 0; j < 8; ++j) {
#pragma unroll
      for (int mask = 1; mask <= 8; mask <<= 1)
        rs[j] += __shfl_xor(rs[j], mask, 32);
      lrow[j] = lrow[j] * al[j] + rs[j];
    }
#pragma unroll
    for (int dt = 0; dt < 4; ++dt)
#pragma unroll
      for (int j = 0; j < 8; ++j) oacc[dt][j] *= al[j];

    // ---- O += P * V  (re-layout P via LDS into A-fragments) --------------
    bf16x16 pf[2];
#pragma unroll
    for (int c = 0; c < 2; ++c) {
      const bf16* p = &pbuf[l16 * KSTR + c * 32 + 8 * half];
      pf[c] = bfcomb(*(const bf16x8*)p, *(const bf16x8*)(p + 16));
    }
#pragma unroll
    for (int dt = 0; dt < 4; ++dt) {
#pragma unroll
      for (int c = 0; c < 2; ++c) {
        const bf16* p = &lds_v[(dt * 16 + l16) * KSTR + c * 32 + 8 * half];
        const bf16x16 bv = bfcomb(*(const bf16x8*)p, *(const bf16x8*)(p + 16));
        oacc[dt] = wmma_bf16(pf[c], bv, oacc[dt]);
      }
    }
  }

  // ---- normalize and store context (bf16, [B,S,HID]) ---------------------
#pragma unroll
  for (int dt = 0; dt < 4; ++dt) {
#pragma unroll
    for (int j = 0; j < 8; ++j) {
      const int s = q0 + j + 8 * half;
      const int d = dt * 16 + l16;
      Ctx[bh + (size_t)s * HID + d] = (bf16)(oacc[dt][j] / lrow[j]);
    }
  }
}

// ---------------------------------------------------------------------------
// Host launcher
// ---------------------------------------------------------------------------
extern "C" void kernel_launch(void* const* d_in, const int* in_sizes, int n_in,
                              void* d_out, int out_size, void* d_ws, size_t ws_size,
                              hipStream_t stream) {
  const float* q  = (const float*)d_in[0];
  const float* k  = (const float*)d_in[1];
  const float* v  = (const float*)d_in[2];
  const float* wq = (const float*)d_in[3];
  const float* bq = (const float*)d_in[4];
  const float* wk = (const float*)d_in[5];
  const float* bk = (const float*)d_in[6];
  const float* wv = (const float*)d_in[7];
  const float* bv = (const float*)d_in[8];
  const float* wo = (const float*)d_in[9];
  const float* bo = (const float*)d_in[10];

  const int M = BATCH * SEQ;            // 8192
  const size_t MN = (size_t)M * HID;    // 8M elements
  bf16* Qb = (bf16*)d_ws;               // 16 MB each; 64 MB total workspace
  bf16* Kb = Qb + MN;
  bf16* Vb = Kb + MN;                   // stored transposed: [B,H,D,S]
  bf16* Cb = Vb + MN;

  dim3 gg(M / 128, HID / 128);          // 64 x 8 blocks
  gemm_bias_kernel<false, 1><<<gg, 256, 0, stream>>>(q, wq, bq, Qb, M, HID, HID);
  gemm_bias_kernel<false, 1><<<gg, 256, 0, stream>>>(k, wk, bk, Kb, M, HID, HID);
  gemm_bias_kernel<false, 2><<<gg, 256, 0, stream>>>(v, wv, bv, Vb, M, HID, HID);

  dim3 ga(SEQ / 128, NHEAD, BATCH);     // 16 x 16 x 4 blocks
  attn_kernel<<<ga, 256, 0, stream>>>(Qb, Kb, Vb, Cb);

  gemm_bias_kernel<true, 0><<<gg, 256, 0, stream>>>(Cb, wo, bo, (float*)d_out,
                                                    M, HID, HID);
}
